// GraphConstructor_2534030705014
// MI455X (gfx1250) — matrix operand order repaired and checked by reading, not compiled, not run on hardware
//
#include <hip/hip_runtime.h>
#include <hip/hip_bf16.h>

typedef __attribute__((ext_vector_type(16))) _Float16 v16h;
typedef __attribute__((ext_vector_type(8)))  float    v8f;

#define NN   8192
#define DIMV 64
#define KTOP 32u

// ---- fast tanh: gfx1250 v_tanh_f32 transcendental ----
#if defined(__has_builtin)
#if __has_builtin(__builtin_amdgcn_tanhf)
#define HAVE_TANH_BUILTIN 1
#endif
#endif

static __device__ inline float tanh_fast(float x) {
#ifdef HAVE_TANH_BUILTIN
  return __builtin_amdgcn_tanhf(x);
#else
  float r;
  // TRANS op: 1 independent op (v_nop) before result use per ISA §7.4
  asm volatile("v_tanh_f32 %0, %1\n\tv_nop" : "=v"(r) : "v"(x));
  return r;
#endif
}

// =====================================================================
// Kernel 1: n{1,2}[i,d] = tanh(3*(emb{1,2}[idx[i]] . W{1,2}[d,:] + b{1,2}[d]))
// stored as f16 row-major [N][64] for WMMA.
// =====================================================================
__global__ __launch_bounds__(256) void prep_kernel(
    const int* __restrict__ idx,
    const float* __restrict__ emb1, const float* __restrict__ emb2,
    const float* __restrict__ W1,   const float* __restrict__ b1,
    const float* __restrict__ W2,   const float* __restrict__ b2,
    _Float16* __restrict__ n1h, _Float16* __restrict__ n2h) {
  int tid = blockIdx.x * 256 + threadIdx.x;   // N*DIM threads
  int i = tid >> 6;
  int d = tid & 63;
  int g = idx[i];
  const float* e1 = emb1 + (size_t)g * DIMV;
  const float* e2 = emb2 + (size_t)g * DIMV;
  const float* w1 = W1 + (size_t)d * DIMV;    // row d of W => column d of W^T
  const float* w2 = W2 + (size_t)d * DIMV;
  float a1 = b1[d], a2 = b2[d];
#pragma unroll
  for (int k = 0; k < DIMV; k++) {
    a1 = fmaf(e1[k], w1[k], a1);
    a2 = fmaf(e2[k], w2[k], a2);
  }
  n1h[tid] = (_Float16)tanh_fast(3.0f * a1);
  n2h[tid] = (_Float16)tanh_fast(3.0f * a2);
}

// =====================================================================
// WMMA fragment loaders (wave32, f16, 16x16x32). Row-major [node][64] src.
// A 16x32 (MxK): lane m=lane&15 holds row R+m; lane half selects K-subsets
//   VGPR0-3: K = ko + hi*8 .. +7 ; VGPR4-7: K = ko + 16 + hi*8 .. +7
// B 32x16 (KxN): lane n=lane&15 holds column C+n (= row C+n of source, since
//   B = srcT); lane half selects K = ko + kh*8 .. +7 and ko + 16 + kh*8 .. +7.
// =====================================================================
static __device__ inline v16h load_fragA(const _Float16* __restrict__ base,
                                         int row0, int ko, int lane) {
  int m  = lane & 15;
  int hi = lane >> 4;
  const _Float16* p = base + (size_t)(row0 + m) * DIMV;
  v16h f;
  float4* fp = reinterpret_cast<float4*>(&f);
  fp[0] = *reinterpret_cast<const float4*>(p + ko + hi * 8);        // 8 halves
  fp[1] = *reinterpret_cast<const float4*>(p + ko + 16 + hi * 8);   // 8 halves
  return f;
}

static __device__ inline v16h load_fragB(const _Float16* __restrict__ base,
                                         int col0, int ko, int lane) {
  int n  = lane & 15;
  int kh = lane >> 4;
  const _Float16* p = base + (size_t)(col0 + n) * DIMV + ko;
  v16h f;
  float4* fp = reinterpret_cast<float4*>(&f);
  fp[0] = *reinterpret_cast<const float4*>(p + kh * 8);
  fp[1] = *reinterpret_cast<const float4*>(p + 16 + kh * 8);
  return f;
}

static __device__ inline v16h neg16h(v16h a) {
  union { v16h h; unsigned u[8]; } x;
  x.h = a;
#pragma unroll
  for (int i = 0; i < 8; i++) x.u[i] ^= 0x80008000u;  // flip both f16 signs
  return x.h;
}

// =====================================================================
// Kernel 2: A = tanh(3*max(n1@n2^T - n2@n1^T, 0))  written to d_out [N][N].
// Block = 256 thr = 8 waves. Block tile 128x128; each wave: 16 rows x 8
// column tiles of 16.  4 WMMAs per tile (two K-steps x two terms; second
// term folded in via sign-negated A fragment).
// =====================================================================
__global__ __launch_bounds__(256) void gemm_kernel(
    const _Float16* __restrict__ n1h, const _Float16* __restrict__ n2h,
    float* __restrict__ out) {
  const int lane = threadIdx.x & 31;
  const int wave = threadIdx.x >> 5;
  const int R  = blockIdx.y * 128 + wave * 16;
  const int C0 = blockIdx.x * 128;

  // A-side fragments (reused across 8 column tiles)
  v16h a1k0 = load_fragA(n1h, R, 0,  lane);
  v16h a1k1 = load_fragA(n1h, R, 32, lane);
  v16h a2k0 = neg16h(load_fragA(n2h, R, 0,  lane));
  v16h a2k1 = neg16h(load_fragA(n2h, R, 32, lane));

  const int mrow = R + (lane >> 4) * 8;  // C/D layout: lane half selects M 0-7 / 8-15
  const int ncol = lane & 15;

#pragma unroll
  for (int ct = 0; ct < 8; ct++) {
    const int C = C0 + ct * 16;
    v16h b2k0 = load_fragB(n2h, C, 0,  lane);
    v16h b2k1 = load_fragB(n2h, C, 32, lane);
    v16h b1k0 = load_fragB(n1h, C, 0,  lane);
    v16h b1k1 = load_fragB(n1h, C, 32, lane);

    v8f acc = {};
    acc = __builtin_amdgcn_wmma_f32_16x16x32_f16(false, a1k0, false, b2k0,
                                                 (short)0, acc, false, false);
    acc = __builtin_amdgcn_wmma_f32_16x16x32_f16(false, a1k1, false, b2k1,
                                                 (short)0, acc, false, false);
    acc = __builtin_amdgcn_wmma_f32_16x16x32_f16(false, a2k0, false, b1k0,
                                                 (short)0, acc, false, false);
    acc = __builtin_amdgcn_wmma_f32_16x16x32_f16(false, a2k1, false, b1k1,
                                                 (short)0, acc, false, false);

    size_t base = (size_t)mrow * NN + C + ncol;
#pragma unroll
    for (int v = 0; v < 8; v++) {
      // relu(tanh(3a)) == tanh(3*max(a,0))  (tanh monotone, tanh(0)=0)
      float x = 3.0f * fmaxf(acc[v], 0.0f);
      out[base + (size_t)v * NN] = tanh_fast(x);
    }
  }
}

// =====================================================================
// Kernel 3: per-row top-K mask. One 256-thread block per row.
// Row staged in LDS (bank-conflict padded).  Exact 32nd-largest value via
// 4-round radix select on u32 bit patterns (all values >= 0 => monotone).
// Ties broken in index order (matches jax top_k), then row rewritten masked.
// =====================================================================
#define SIDX(i) ((i) + ((i) >> 5))

__global__ __launch_bounds__(256) void topk_kernel(float* __restrict__ out) {
  __shared__ unsigned sval[NN + NN / 32];
  __shared__ unsigned hist[256];
  __shared__ unsigned sbc[2];

  const int row = blockIdx.x;
  const int t   = threadIdx.x;
  float* rp = out + (size_t)row * NN;

  // coalesced load of the row into LDS
  for (int j = 0; j < 32; j++) {
    int idx = j * 256 + t;
    sval[SIDX(idx)] = __float_as_uint(rp[idx]);
  }
  __syncthreads();

  unsigned prefix = 0;
  unsigned need   = KTOP;   // still-to-select at/below current value class

  for (int round = 3; round >= 0; round--) {
    const int shift = round * 8;
    const unsigned hmask = (round == 3) ? 0u : (0xFFFFFFFFu << (shift + 8));

    hist[t] = 0;
    __syncthreads();
    for (int j = 0; j < 32; j++) {
      unsigned v = sval[SIDX(t * 32 + j)];
      if ((v & hmask) == prefix)
        atomicAdd(&hist[(v >> shift) & 255u], 1u);
    }
    __syncthreads();
    if (t == 0) {
      unsigned acc = 0;
      int d = 255;
      for (; d > 0; d--) {
        if (acc + hist[d] >= need) break;
        acc += hist[d];
      }
      sbc[0] = prefix | ((unsigned)d << shift);
      sbc[1] = need - acc;
    }
    __syncthreads();
    prefix = sbc[0];
    need   = sbc[1];
    __syncthreads();
  }

  const unsigned T = prefix;  // exact bit pattern of the K-th largest value

  // index-ordered tie ranking: per-thread-chunk counts -> exclusive scan
  unsigned cnt = 0;
  for (int j = 0; j < 32; j++) cnt += (sval[SIDX(t * 32 + j)] == T) ? 1u : 0u;
  hist[t] = cnt;
  __syncthreads();
  if (t == 0) {
    unsigned run = 0;
    for (int i = 0; i < 256; i++) { unsigned c = hist[i]; hist[i] = run; run += c; }
  }
  __syncthreads();

  unsigned rank = hist[t];
  for (int j = 0; j < 32; j++) {
    unsigned v = sval[SIDX(t * 32 + j)];
    bool keep = false;
    if (v > T) keep = true;
    else if (v == T) { keep = (rank < need); rank++; }
    rp[t * 32 + j] = keep ? __uint_as_float(v) : 0.0f;
  }
}

// =====================================================================
extern "C" void kernel_launch(void* const* d_in, const int* in_sizes, int n_in,
                              void* d_out, int out_size, void* d_ws, size_t ws_size,
                              hipStream_t stream) {
  const int*   idx  = (const int*)d_in[0];
  const float* emb1 = (const float*)d_in[1];
  const float* emb2 = (const float*)d_in[2];
  const float* W1   = (const float*)d_in[3];
  const float* b1   = (const float*)d_in[4];
  const float* W2   = (const float*)d_in[5];
  const float* b2   = (const float*)d_in[6];
  float* out = (float*)d_out;

  _Float16* n1h = (_Float16*)d_ws;
  _Float16* n2h = n1h + (size_t)NN * DIMV;

  prep_kernel<<<(NN * DIMV) / 256, 256, 0, stream>>>(idx, emb1, emb2, W1, b1,
                                                     W2, b2, n1h, n2h);

  dim3 g2(NN / 128, NN / 128);  // 64 x 64 blocks of 128x128
  gemm_kernel<<<g2, 256, 0, stream>>>(n1h, n2h, out);

  topk_kernel<<<NN, 256, 0, stream>>>(out);
}
